// GlobalSparseAttention_1623497638387
// MI455X (gfx1250) — compile-verified, matrix operand order checked by reading, not run on hardware
//
#include <hip/hip_runtime.h>
#include <hip/hip_bf16.h>

typedef __attribute__((ext_vector_type(16))) _Float16 v16h;
typedef __attribute__((ext_vector_type(8)))  _Float16 v8h;
typedef __attribute__((ext_vector_type(8)))  float    v8f;

#define NTOK   4096
#define CDIM   1024
#define NHEAD  16
#define DHEAD  64
#define NFRM   16

static __device__ __forceinline__ v8f wmma_f16(v16h a, v16h b, v8f c) {
    return __builtin_amdgcn_wmma_f32_16x16x32_f16(false, a, false, b, (short)0, c, false, false);
}

// Load a 16(row) x 32(K) f16 fragment from a row-major matrix (row stride = ld halfs).
// Matches CDNA5 A/B fragment layout for v_wmma_*_16x16x32_f16:
//   lane<16 : row=lane,    halves 0..7 = K 0..7,  halves 8..15 = K 16..23
//   lane>=16: row=lane-16, halves 0..7 = K 8..15, halves 8..15 = K 24..31
static __device__ __forceinline__ v16h load_frag_g(const _Float16* base, int ld) {
    int lane = threadIdx.x & 31;
    int row  = lane & 15;
    int koff = (lane >> 4) << 3;           // 0 or 8
    const _Float16* p = base + (size_t)row * ld + koff;
    v8h a = *(const v8h*)(p);
    v8h b = *(const v8h*)(p + 16);
    v16h f;
#pragma unroll
    for (int i = 0; i < 8; ++i) { f[i] = a[i]; f[8 + i] = b[i]; }
    return f;
}

// Same fragment gather, but from LDS (addrspace(3) -> ds_load_b128).
static __device__ __forceinline__ v16h load_frag_l(const _Float16* base, int ld) {
    int lane = threadIdx.x & 31;
    int row  = lane & 15;
    int koff = (lane >> 4) << 3;
    const _Float16* p = base + row * ld + koff;
    v8h a = *(const v8h*)(p);
    v8h b = *(const v8h*)(p + 16);
    v16h f;
#pragma unroll
    for (int i = 0; i < 8; ++i) { f[i] = a[i]; f[8 + i] = b[i]; }
    return f;
}

// CDNA5 async copy: global -> LDS, 16 bytes per lane, tracked by ASYNCcnt.
static __device__ __forceinline__ void async_b128(unsigned lds_off, const void* g) {
    asm volatile("global_load_async_to_lds_b128 %0, %1, off"
                 :: "v"(lds_off), "v"(g) : "memory");
}

static __device__ __forceinline__ float redmax16(float v) {
    v = fmaxf(v, __shfl_xor(v, 1, 32));
    v = fmaxf(v, __shfl_xor(v, 2, 32));
    v = fmaxf(v, __shfl_xor(v, 4, 32));
    v = fmaxf(v, __shfl_xor(v, 8, 32));
    return v;
}
static __device__ __forceinline__ float redsum16(float v) {
    v += __shfl_xor(v, 1, 32);
    v += __shfl_xor(v, 2, 32);
    v += __shfl_xor(v, 4, 32);
    v += __shfl_xor(v, 8, 32);
    return v;
}

// ---------------------------------------------------------------- convert
// 8 elements per thread: 2x float4 in, 1x 16-B f16 store out. n8 = n/8.
__global__ __launch_bounds__(256) void cvt_f32_f16(const float* __restrict__ src,
                                                   _Float16* __restrict__ dst, int n8) {
    int i = blockIdx.x * 256 + threadIdx.x;
    if (i < n8) {
        const float4* s = (const float4*)src + (size_t)i * 2;
        float4 a = s[0], b = s[1];
        v8h o;
        o[0] = (_Float16)a.x; o[1] = (_Float16)a.y; o[2] = (_Float16)a.z; o[3] = (_Float16)a.w;
        o[4] = (_Float16)b.x; o[5] = (_Float16)b.y; o[6] = (_Float16)b.z; o[7] = (_Float16)b.w;
        ((v8h*)dst)[i] = o;
    }
}

// ---------------------------------------------------------------- QKV GEMM
// qkv[n,m] = sum_c x[n,c]*qkv_w[m,c] + qkv_b[m], m in [0,3072)
// route (wave-uniform in col0): m<1024 -> q[h][n][d]; m<2048 -> k[h][n][d]; else vt[h][d][n]
__global__ __launch_bounds__(128) void qkv_gemm(const _Float16* __restrict__ xh,
                                                const _Float16* __restrict__ wh,
                                                const float* __restrict__ bias,
                                                _Float16* __restrict__ qo,
                                                _Float16* __restrict__ ko,
                                                _Float16* __restrict__ vto) {
    const int wid  = threadIdx.x >> 5;
    const int lane = threadIdx.x & 31;
    const int task = blockIdx.x * 4 + wid;
    const int ct = task % 48;              // 48 col tiles of 64
    const int rt = task / 48;              // 128 row tiles of 32
    if (rt >= 128) return;
    const int row0 = rt * 32;
    const int col0 = ct * 64;

    const _Float16* a0p = xh + (size_t)row0 * CDIM;
    const _Float16* a1p = xh + (size_t)(row0 + 16) * CDIM;
    const _Float16* bp[4];
#pragma unroll
    for (int j = 0; j < 4; ++j) bp[j] = wh + (size_t)(col0 + 16 * j) * CDIM;

    v8f acc[2][4] = {};
    for (int kk = 0; kk < CDIM; kk += 32) {
        __builtin_prefetch(a0p + kk + 128, 0, 3);
        __builtin_prefetch(bp[0] + kk + 128, 0, 3);
        v16h a0 = load_frag_g(a0p + kk, CDIM);
        v16h a1 = load_frag_g(a1p + kk, CDIM);
        v16h b[4];
#pragma unroll
        for (int j = 0; j < 4; ++j) b[j] = load_frag_g(bp[j] + kk, CDIM);
#pragma unroll
        for (int j = 0; j < 4; ++j) {
            acc[0][j] = wmma_f16(a0, b[j], acc[0][j]);
            acc[1][j] = wmma_f16(a1, b[j], acc[1][j]);
        }
    }

    const int half = lane >> 4;
    const int nlo  = lane & 15;
    float bj[4];
#pragma unroll
    for (int j = 0; j < 4; ++j) bj[j] = bias[col0 + 16 * j + nlo];

    const int mh = ((col0 & (CDIM - 1)) >> 6);     // head (uniform)
    if (col0 < CDIM) {                             // ---- Q: [h][n][d]
        _Float16* dst = qo + (size_t)mh * NTOK * DHEAD;
#pragma unroll
        for (int i = 0; i < 2; ++i)
#pragma unroll
            for (int j = 0; j < 4; ++j)
#pragma unroll
                for (int r = 0; r < 8; ++r) {
                    int n = row0 + i * 16 + r + 8 * half;
                    int d = ((col0 + j * 16) & 63) + nlo;
                    dst[(size_t)n * DHEAD + d] = (_Float16)(acc[i][j][r] + bj[j]);
                }
    } else if (col0 < 2 * CDIM) {                  // ---- K: [h][n][d]
        _Float16* dst = ko + (size_t)mh * NTOK * DHEAD;
#pragma unroll
        for (int i = 0; i < 2; ++i)
#pragma unroll
            for (int j = 0; j < 4; ++j)
#pragma unroll
                for (int r = 0; r < 8; ++r) {
                    int n = row0 + i * 16 + r + 8 * half;
                    int d = ((col0 + j * 16) & 63) + nlo;
                    dst[(size_t)n * DHEAD + d] = (_Float16)(acc[i][j][r] + bj[j]);
                }
    } else {                                       // ---- V^T: [h][d][n]
        _Float16* dst = vto + (size_t)mh * DHEAD * NTOK;
#pragma unroll
        for (int i = 0; i < 2; ++i)
#pragma unroll
            for (int j = 0; j < 4; ++j)
#pragma unroll
                for (int r = 0; r < 8; ++r) {
                    int n = row0 + i * 16 + r + 8 * half;
                    int d = ((col0 + j * 16) & 63) + nlo;
                    dst[(size_t)d * NTOK + n] = (_Float16)(acc[i][j][r] + bj[j]);
                }
    }
}

// ---------------------------------------------------------------- attention
// Block = 4 waves, all on the SAME head, 4 consecutive 16-row query tiles.
// K (32x64) and V^T (64x32) tiles are staged cooperatively into LDS with
// double-buffered global_load_async_to_lds_b128 and consumed by all 4 waves.
__global__ __launch_bounds__(128) void attn_kernel(const _Float16* __restrict__ qm,
                                                   const _Float16* __restrict__ km,
                                                   const _Float16* __restrict__ vtm,
                                                   const int* __restrict__ frame_ids,
                                                   const unsigned char* __restrict__ is_hub,
                                                   const unsigned char* __restrict__ adj,
                                                   _Float16* __restrict__ attnout) {
    __shared__ __align__(16) _Float16 ldsK[2][32 * DHEAD];    // 8 KB
    __shared__ __align__(16) _Float16 ldsVT[2][DHEAD * 32];   // 8 KB
    __shared__ __align__(16) _Float16 ldsP[4][16 * 32];       // 4 KB
    const int wid  = threadIdx.x >> 5;
    const int lane = threadIdx.x & 31;
    const int t    = threadIdx.x;                   // 0..127
    const int task = blockIdx.x * 4 + wid;          // 4096 tasks
    const int h  = task >> 8;                       // head (uniform across block)
    const int qt = task & 255;                      // 16-row query tile
    const int q0 = qt * 16;
    const int half = lane >> 4;
    const int nlo  = lane & 15;

    const _Float16* qbase = qm + ((size_t)h * NTOK + q0) * DHEAD;
    v16h qa0 = load_frag_g(qbase +  0, DHEAD);
    v16h qa1 = load_frag_g(qbase + 32, DHEAD);

    // per-row metadata (rows this lane participates in: q0 + r + 8*half)
    int fiq[8]; int hubq[8]; unsigned arow[8];
#pragma unroll
    for (int r = 0; r < 8; ++r) {
        int qr = q0 + r + 8 * half;
        fiq[r]  = frame_ids[qr];
        hubq[r] = is_hub[qr];
        unsigned m = 0;
        for (int j = 0; j < NFRM; ++j)
            m |= (unsigned)(adj[fiq[r] * NFRM + j] != 0) << j;
        arow[r] = m;
    }

    // global bases for async staging (head-uniform)
    const char* kgbase = (const char*)(km + (size_t)h * NTOK * DHEAD);
    const char* vgbase = (const char*)(vtm + (size_t)h * DHEAD * NTOK);
    const unsigned kl0 = (unsigned)(uintptr_t)&ldsK[0][0];
    const unsigned kl1 = (unsigned)(uintptr_t)&ldsK[1][0];
    const unsigned vl0 = (unsigned)(uintptr_t)&ldsVT[0][0];
    const unsigned vl1 = (unsigned)(uintptr_t)&ldsVT[1][0];

    // stage(buf, kb): K tile = 4 KB contiguous; VT tile = 64 rows x 64 B.
    auto stage = [&](int buf, int kb) {
        unsigned kl = buf ? kl1 : kl0;
        unsigned vl = buf ? vl1 : vl0;
        const char* kg = kgbase + (size_t)kb * DHEAD * 2;     // contiguous 4 KB
        async_b128(kl + t * 16,        kg + t * 16);
        async_b128(kl + t * 16 + 2048, kg + t * 16 + 2048);
        const char* vg = vgbase + (size_t)kb * 2;             // row stride NTOK*2 B
        int c0 = t, c1 = t + 128;                              // 256 16-B chunks
        async_b128(vl + (c0 >> 2) * 64 + (c0 & 3) * 16,
                   vg + (size_t)(c0 >> 2) * (NTOK * 2) + (c0 & 3) * 16);
        async_b128(vl + (c1 >> 2) * 64 + (c1 & 3) * 16,
                   vg + (size_t)(c1 >> 2) * (NTOK * 2) + (c1 & 3) * 16);
    };

    const float scale = 0.125f;                     // 1/sqrt(64)
    const float L2E   = 1.44269504088896340736f;
    float mrow[8], lrow[8];
    v8f O[4] = {};
#pragma unroll
    for (int r = 0; r < 8; ++r) { mrow[r] = -1e30f; lrow[r] = 0.0f; }

    stage(0, 0);                                    // prologue
    for (int kb = 0, it = 0; kb < NTOK; kb += 32, ++it) {
        const int cur = it & 1;
        // my cur-buffer copies complete:
        asm volatile("s_wait_asynccnt 0x0" ::: "memory");
        // everyone's copies complete; also fences last iter's LDS reads
        // before we overwrite that buffer below:
        __syncthreads();
        if (kb + 32 < NTOK) stage(cur ^ 1, kb + 32);

        v16h kb0d0 = load_frag_l(&ldsK[cur][0],            DHEAD);
        v16h kb0d1 = load_frag_l(&ldsK[cur][32],           DHEAD);
        v16h kb1d0 = load_frag_l(&ldsK[cur][16 * DHEAD],   DHEAD);
        v16h kb1d1 = load_frag_l(&ldsK[cur][16 * DHEAD + 32], DHEAD);
        v8f S0 = {}, S1 = {};
        S0 = wmma_f16(qa0, kb0d0, S0);
        S0 = wmma_f16(qa1, kb0d1, S0);
        S1 = wmma_f16(qa0, kb1d0, S1);
        S1 = wmma_f16(qa1, kb1d1, S1);

        // mask + scale (score (row, key): row = r+8*half, key col = nlo / nlo+16)
        int k0 = kb + nlo, k1 = kb + 16 + nlo;
        int fk0 = frame_ids[k0], fk1 = frame_ids[k1];
        int hk0 = is_hub[k0],    hk1 = is_hub[k1];
        float s0f[8], s1f[8];
        bool aok0[8], aok1[8];
#pragma unroll
        for (int r = 0; r < 8; ++r) {
            aok0[r] = (((arow[r] >> fk0) & 1u) != 0u) &&
                      ((fiq[r] == fk0) || (!hubq[r] && !hk0));
            aok1[r] = (((arow[r] >> fk1) & 1u) != 0u) &&
                      ((fiq[r] == fk1) || (!hubq[r] && !hk1));
            s0f[r] = aok0[r] ? S0[r] * scale : -1e30f;
            s1f[r] = aok1[r] ? S1[r] * scale : -1e30f;
        }

        float alpha[8];
#pragma unroll
        for (int r = 0; r < 8; ++r) {
            float mb = redmax16(fmaxf(s0f[r], s1f[r]));
            float mnew = fmaxf(mrow[r], mb);
            alpha[r] = exp2f((mrow[r] - mnew) * L2E);
            float p0 = aok0[r] ? exp2f((s0f[r] - mnew) * L2E) : 0.0f;
            float p1 = aok1[r] ? exp2f((s1f[r] - mnew) * L2E) : 0.0f;
            lrow[r] = lrow[r] * alpha[r] + redsum16(p0 + p1);
            mrow[r] = mnew;
            ldsP[wid][(r + 8 * half) * 32 + nlo]      = (_Float16)p0;
            ldsP[wid][(r + 8 * half) * 32 + nlo + 16] = (_Float16)p1;
        }
#pragma unroll
        for (int j = 0; j < 4; ++j)
#pragma unroll
            for (int r = 0; r < 8; ++r) O[j][r] *= alpha[r];

        // wave-local LDS bounce: C-layout -> A-fragment layout
        asm volatile("s_wait_dscnt 0x0" ::: "memory");
        v16h pf = load_frag_l(&ldsP[wid][0], 32);
#pragma unroll
        for (int j = 0; j < 4; ++j) {
            v16h bf = load_frag_l(&ldsVT[cur][j * 16 * 32], 32);
            O[j] = wmma_f16(pf, bf, O[j]);
        }
    }

    float inv[8];
#pragma unroll
    for (int r = 0; r < 8; ++r) inv[r] = 1.0f / lrow[r];
#pragma unroll
    for (int j = 0; j < 4; ++j)
#pragma unroll
        for (int r = 0; r < 8; ++r) {
            int n = q0 + r + 8 * half;
            int d = j * 16 + nlo;
            attnout[(size_t)n * CDIM + h * DHEAD + d] = (_Float16)(O[j][r] * inv[r]);
        }
}

// ---------------------------------------------------------------- projection GEMM
__global__ __launch_bounds__(128) void proj_gemm(const _Float16* __restrict__ ah,
                                                 const _Float16* __restrict__ wh,
                                                 const float* __restrict__ bias,
                                                 float* __restrict__ out) {
    const int wid  = threadIdx.x >> 5;
    const int lane = threadIdx.x & 31;
    const int task = blockIdx.x * 4 + wid;
    const int ct = task % 16;              // 16 col tiles of 64
    const int rt = task / 16;              // 128 row tiles of 32
    if (rt >= 128) return;
    const int row0 = rt * 32;
    const int col0 = ct * 64;

    const _Float16* a0p = ah + (size_t)row0 * CDIM;
    const _Float16* a1p = ah + (size_t)(row0 + 16) * CDIM;
    const _Float16* bp[4];
#pragma unroll
    for (int j = 0; j < 4; ++j) bp[j] = wh + (size_t)(col0 + 16 * j) * CDIM;

    v8f acc[2][4] = {};
    for (int kk = 0; kk < CDIM; kk += 32) {
        __builtin_prefetch(a0p + kk + 128, 0, 3);
        v16h a0 = load_frag_g(a0p + kk, CDIM);
        v16h a1 = load_frag_g(a1p + kk, CDIM);
        v16h b[4];
#pragma unroll
        for (int j = 0; j < 4; ++j) b[j] = load_frag_g(bp[j] + kk, CDIM);
#pragma unroll
        for (int j = 0; j < 4; ++j) {
            acc[0][j] = wmma_f16(a0, b[j], acc[0][j]);
            acc[1][j] = wmma_f16(a1, b[j], acc[1][j]);
        }
    }

    const int half = lane >> 4;
    const int nlo  = lane & 15;
    float bj[4];
#pragma unroll
    for (int j = 0; j < 4; ++j) bj[j] = bias[col0 + 16 * j + nlo];
#pragma unroll
    for (int i = 0; i < 2; ++i)
#pragma unroll
        for (int j = 0; j < 4; ++j)
#pragma unroll
            for (int r = 0; r < 8; ++r) {
                int n = row0 + i * 16 + r + 8 * half;
                int m = col0 + j * 16 + nlo;
                out[(size_t)n * CDIM + m] = acc[i][j][r] + bj[j];
            }
}

// ---------------------------------------------------------------- launch
extern "C" void kernel_launch(void* const* d_in, const int* in_sizes, int n_in,
                              void* d_out, int out_size, void* d_ws, size_t ws_size,
                              hipStream_t stream) {
    const float* x       = (const float*)d_in[0];
    const float* qkv_w   = (const float*)d_in[1];
    const float* qkv_b   = (const float*)d_in[2];
    const float* proj_w  = (const float*)d_in[3];
    const float* proj_b  = (const float*)d_in[4];
    const unsigned char* adj      = (const unsigned char*)d_in[5];
    const int*           frame_id = (const int*)d_in[6];
    const unsigned char* is_hub   = (const unsigned char*)d_in[7];
    float* out = (float*)d_out;

    char* ws = (char*)d_ws;
    // workspace carve (bytes)
    _Float16* xh  = (_Float16*)(ws + 0);            //  8 MB: x in f16 (reused as attnout later)
    _Float16* wh  = (_Float16*)(ws + 8388608);      //  6 MB: qkv_w f16
    _Float16* pwh = (_Float16*)(ws + 14680064);     //  2 MB: proj_w f16
    _Float16* qh  = (_Float16*)(ws + 16777216);     //  8 MB: q [h][n][d]
    _Float16* kh  = (_Float16*)(ws + 25165824);     //  8 MB: k [h][n][d]
    _Float16* vth = (_Float16*)(ws + 33554432);     //  8 MB: v^T [h][d][n]
    _Float16* ao  = xh;                             //  attention out reuses xh region

    // 1) convert inputs to f16 (8 elems/thread)
    {
        int n1 = (NTOK * CDIM) / 8;
        cvt_f32_f16<<<(n1 + 255) / 256, 256, 0, stream>>>(x, xh, n1);
        int n2 = (3 * CDIM * CDIM) / 8;
        cvt_f32_f16<<<(n2 + 255) / 256, 256, 0, stream>>>(qkv_w, wh, n2);
        int n3 = (CDIM * CDIM) / 8;
        cvt_f32_f16<<<(n3 + 255) / 256, 256, 0, stream>>>(proj_w, pwh, n3);
    }
    // 2) QKV projection (WMMA), routed into q / k / v^T per-head layouts
    qkv_gemm<<<(128 * 48) / 4, 128, 0, stream>>>(xh, wh, qkv_b, qh, kh, vth);
    // 3) masked flash attention (WMMA + async LDS staging)
    attn_kernel<<<(NHEAD * 256) / 4, 128, 0, stream>>>(qh, kh, vth, frame_id, is_hub, adj, ao);
    // 4) output projection (WMMA), f32 out + bias
    proj_gemm<<<(128 * 16) / 4, 128, 0, stream>>>(ao, pwh, proj_b, out);
}
